// DeltaIrisTokenizer_33904471835541
// MI455X (gfx1250) — matrix operand order repaired and checked by reading, not compiled, not run on hardware
//
#include <hip/hip_runtime.h>
#include <cstdint>
#include <cstddef>

// ---- problem constants ----
#define B_    8
#define S_    32
#define P_    64
#define D_    256
#define H_    1024
#define KCB_  1024
#define OBS_  64
#define CTX_  4
#define NH_   4
#define DH_   64

typedef __attribute__((ext_vector_type(16))) __bf16 v16bf;
typedef __attribute__((ext_vector_type(8)))  float  v8f;
typedef unsigned short u16;

struct Frag32B { uint4 lo, hi; };

__device__ __forceinline__ float bf2f(u16 h) {
  unsigned u = ((unsigned)h) << 16;
  return __builtin_bit_cast(float, u);
}
__device__ __forceinline__ u16 f2bf(float f) {
  unsigned u = __builtin_bit_cast(unsigned, f);
  unsigned r = u + 0x7FFFu + ((u >> 16) & 1u);   // round-to-nearest-even
  return (u16)(r >> 16);
}

// ---- CDNA5 async global->LDS staging (guarded; falls back to sync copy) ----
#if __has_builtin(__builtin_amdgcn_global_load_async_to_lds_b128)
#define HAS_ASYNC_LDS 1
typedef int v4i_vs __attribute__((vector_size(16)));
typedef __attribute__((address_space(1))) v4i_vs* as1_v4i;   // global int4*
typedef __attribute__((address_space(3))) v4i_vs* as3_v4i;   // LDS int4*
#else
#define HAS_ASYNC_LDS 0
#endif

__device__ __forceinline__ void async_ld16(const void* g, void* l) {
#if HAS_ASYNC_LDS
  // low 32 bits of a generic shared pointer are the LDS offset (aperture truncation)
  __builtin_amdgcn_global_load_async_to_lds_b128(
      (as1_v4i)(uintptr_t)g,
      (as3_v4i)(unsigned)(uintptr_t)l,
      0, 0);
#else
  *(uint4*)l = *(const uint4*)g;
#endif
}

template <int N>
__device__ __forceinline__ void wait_async_lds() {
#if __has_builtin(__builtin_amdgcn_s_wait_asynccnt)
  __builtin_amdgcn_s_wait_asynccnt(N);
#elif HAS_ASYNC_LDS
  asm volatile("s_wait_asynccnt %0" :: "i"(N) : "memory");
#endif
}

// ---- WMMA fragment loaders (bf16, per CDNA5 ISA 7.12.2 layouts) ----
// A 16x32 bf16: lanes 0-15 rows M, elems 0..7 = K[half*8..], elems 8..15 = K[16+half*8..]
__device__ __forceinline__ v16bf load_frag_a(const u16* __restrict__ A, int lda,
                                             int m0, int k0, int lane15, int half) {
  const u16* p = A + (size_t)(m0 + lane15) * lda + k0 + half * 8;
  Frag32B f;
  f.lo = *(const uint4*)(p);        // K = k0+half*8 .. +7
  f.hi = *(const uint4*)(p + 16);   // K = k0+16+half*8 .. +7
  return __builtin_bit_cast(v16bf, f);
}
// B fragment from LDS-staged tile: tile layout [n_local][32 k], lane's N = j*16+lane15
__device__ __forceinline__ v16bf load_frag_b_lds(const u16* tile, int j, int lane15, int half) {
  const u16* p = tile + (size_t)(j * 16 + lane15) * 32 + half * 8;
  Frag32B f;
  f.lo = *(const uint4*)(p);
  f.hi = *(const uint4*)(p + 16);
  return __builtin_bit_cast(v16bf, f);
}

__device__ __forceinline__ void store_tile(v8f acc, int row0, int col0,
    const float* __restrict__ bias, float* __restrict__ outF, u16* __restrict__ outB,
    int ldc, int relu, int rpg, long long gStride, long long rOff,
    int lane15, int half) {
  int col = col0 + lane15;
  float bv = bias ? bias[col] : 0.f;
#pragma unroll
  for (int r = 0; r < 8; ++r) {
    int row = row0 + half * 8 + r;          // C/D layout: vgpr r -> M=r (+8 for lanes 16-31)
    float v = acc[r] + bv;
    if (relu) v = fmaxf(v, 0.f);
    long long orow = (long long)(row / rpg) * gStride + rOff + (row % rpg);
    size_t o = (size_t)orow * (size_t)ldc + (size_t)col;
    if (outF) outF[o] = v;
    if (outB) outB[o] = f2bf(v);
  }
}

// ---- WMMA GEMM: C[M,N] = A[M,K](bf16) * W[N,K]^T(bf16) + bias, opt ReLU ----
// Block = 128 threads = 4 waves; block tile 128x64; wave tile 32x64 (2x4 WMMA accs).
// B (weight) 64x32 tile is staged to LDS via async DMA, double-buffered, shared by all waves.
__global__ void __launch_bounds__(128)
k_gemm(const u16* __restrict__ A, int lda,
       const u16* __restrict__ Bm, int ldb,
       const float* __restrict__ bias, int K,
       float* __restrict__ outF, u16* __restrict__ outB,
       int ldc, int relu, int rpg, long long gStride, long long rOff) {
  __shared__ __align__(16) u16 btile[2][64 * 32];   // 2 x 4 KB

  int t      = threadIdx.x;
  int lane   = t & 31;
  int wave   = t >> 5;
  int lane15 = lane & 15;
  int half   = lane >> 4;
  int m0 = blockIdx.y * 128 + wave * 32;
  int n0 = blockIdx.x * 64;

  // staging map: 256 16-byte segments per tile; thread t moves segs 2t and 2t+1
  int s0 = 2 * t, s1 = 2 * t + 1;
  int n_a = s0 >> 2, kpa = (s0 & 3) * 8;   // 8 bf16 per segment
  int n_b = s1 >> 2, kpb = (s1 & 3) * 8;

  v8f acc[2][4] = {};

  // prologue: stage k-tile 0 into buffer 0
  async_ld16(Bm + (size_t)(n0 + n_a) * ldb + 0 + kpa, &btile[0][n_a * 32 + kpa]);
  async_ld16(Bm + (size_t)(n0 + n_b) * ldb + 0 + kpb, &btile[0][n_b * 32 + kpb]);

  int nk = K >> 5;
  for (int ik = 0; ik < nk; ++ik) {
    int kk = ik << 5;
    int cb = ik & 1;
    if (ik + 1 < nk) {
      int kn = kk + 32;
      async_ld16(Bm + (size_t)(n0 + n_a) * ldb + kn + kpa, &btile[cb ^ 1][n_a * 32 + kpa]);
      async_ld16(Bm + (size_t)(n0 + n_b) * ldb + kn + kpb, &btile[cb ^ 1][n_b * 32 + kpb]);
      wait_async_lds<2>();   // wait for current tile only; next tile stays in flight
    } else {
      wait_async_lds<0>();
    }
    __syncthreads();

    v16bf a0 = load_frag_a(A, lda, m0,      kk, lane15, half);
    v16bf a1 = load_frag_a(A, lda, m0 + 16, kk, lane15, half);
    v16bf bf0 = load_frag_b_lds(&btile[cb][0], 0, lane15, half);
    v16bf bf1 = load_frag_b_lds(&btile[cb][0], 1, lane15, half);
    v16bf bf2 = load_frag_b_lds(&btile[cb][0], 2, lane15, half);
    v16bf bf3 = load_frag_b_lds(&btile[cb][0], 3, lane15, half);

    acc[0][0] = __builtin_amdgcn_wmma_f32_16x16x32_bf16(false, a0, false, bf0, (short)0, acc[0][0], false, false);
    acc[0][1] = __builtin_amdgcn_wmma_f32_16x16x32_bf16(false, a0, false, bf1, (short)0, acc[0][1], false, false);
    acc[0][2] = __builtin_amdgcn_wmma_f32_16x16x32_bf16(false, a0, false, bf2, (short)0, acc[0][2], false, false);
    acc[0][3] = __builtin_amdgcn_wmma_f32_16x16x32_bf16(false, a0, false, bf3, (short)0, acc[0][3], false, false);
    acc[1][0] = __builtin_amdgcn_wmma_f32_16x16x32_bf16(false, a1, false, bf0, (short)0, acc[1][0], false, false);
    acc[1][1] = __builtin_amdgcn_wmma_f32_16x16x32_bf16(false, a1, false, bf1, (short)0, acc[1][1], false, false);
    acc[1][2] = __builtin_amdgcn_wmma_f32_16x16x32_bf16(false, a1, false, bf2, (short)0, acc[1][2], false, false);
    acc[1][3] = __builtin_amdgcn_wmma_f32_16x16x32_bf16(false, a1, false, bf3, (short)0, acc[1][3], false, false);

    __syncthreads();   // all waves done reading btile[cb] before it is re-staged
  }

#pragma unroll
  for (int i = 0; i < 2; ++i)
#pragma unroll
    for (int j = 0; j < 4; ++j)
      store_tile(acc[i][j], m0 + i * 16, n0 + j * 16, bias, outF, outB,
                 ldc, relu, rpg, gStride, rOff, lane15, half);
}

// ---- f32 -> bf16 conversion ----
__global__ void k_cvt(u16* __restrict__ dst, const float* __restrict__ src, int n) {
  int i = blockIdx.x * blockDim.x + threadIdx.x;
  if (i < n) dst[i] = f2bf(src[i]);
}

// ---- token expand: tok[(bs,p),d] = tokA[bs,d] + pos[p,d] ----
__global__ void k_expand(float* __restrict__ tok, const float* __restrict__ tokA,
                         const float* __restrict__ pos) {
  int i = blockIdx.x * blockDim.x + threadIdx.x;   // < 16384*256
  int d = i & 255, row = i >> 8;
  int p = row & 63, bs = row >> 6;
  tok[i] = tokA[bs * D_ + d] + pos[p * D_ + d];
}

// ---- flash-style attention on bf16 qkv (row layout [q|k|v], 3*D wide) ----
template <int L>
__global__ void k_attn(const u16* __restrict__ qkv, u16* __restrict__ out,
                       int nSeq, float scale) {
  int t = blockIdx.x * blockDim.x + threadIdx.x;
  if (t >= nSeq * NH_ * L) return;
  int qi  = t % L;
  int h   = (t / L) % NH_;
  int seq = t / (L * NH_);
  const u16* base = qkv + (size_t)seq * L * (3 * D_);
  const u16* qrow = base + (size_t)qi * (3 * D_) + h * DH_;
  float qv[DH_];
#pragma unroll
  for (int d = 0; d < DH_; ++d) qv[d] = bf2f(qrow[d]);
  float m = -3.0e38f, l = 0.f;
  float o[DH_];
#pragma unroll
  for (int d = 0; d < DH_; ++d) o[d] = 0.f;
  for (int k = 0; k < L; ++k) {
    const u16* krow = base + (size_t)k * (3 * D_) + D_ + h * DH_;
    const u16* vrow = base + (size_t)k * (3 * D_) + 2 * D_ + h * DH_;
    float s = 0.f;
#pragma unroll
    for (int d = 0; d < DH_; ++d) s += qv[d] * bf2f(krow[d]);
    s *= scale;
    float nm   = fmaxf(m, s);
    float corr = __expf(m - nm);
    float pw   = __expf(s - nm);
    l = l * corr + pw;
#pragma unroll
    for (int d = 0; d < DH_; ++d) o[d] = o[d] * corr + pw * bf2f(vrow[d]);
    m = nm;
  }
  float inv = 1.f / l;
  u16* orow = out + (size_t)(seq * L + qi) * D_ + h * DH_;
#pragma unroll
  for (int d = 0; d < DH_; ++d) orow[d] = f2bf(o[d] * inv);
}

// ---- residual add + LayerNorm (D=256), one wave per row; emits f32 (+opt bf16) ----
__global__ void k_add_ln(const float* __restrict__ xa, const float* __restrict__ xb,
                         const float* __restrict__ g, const float* __restrict__ b,
                         float* __restrict__ out, u16* __restrict__ outbf, int rows) {
  int wid  = (blockIdx.x * blockDim.x + threadIdx.x) >> 5;
  int lane = threadIdx.x & 31;
  if (wid >= rows) return;
  const float* pa = xa + (size_t)wid * D_;
  const float* pb = xb + (size_t)wid * D_;
  float v[8], s = 0.f, s2 = 0.f;
#pragma unroll
  for (int i = 0; i < 8; ++i) {
    int c = lane + i * 32;
    float x = pa[c] + pb[c];
    v[i] = x; s += x; s2 += x * x;
  }
#pragma unroll
  for (int o = 16; o >= 1; o >>= 1) {
    s  += __shfl_xor(s,  o, 32);
    s2 += __shfl_xor(s2, o, 32);
  }
  float mean = s * (1.f / D_);
  float var  = s2 * (1.f / D_) - mean * mean;
  float inv  = rsqrtf(var + 1e-5f);
#pragma unroll
  for (int i = 0; i < 8; ++i) {
    int c = lane + i * 32;
    float y = (v[i] - mean) * inv * g[c] + b[c];
    out[(size_t)wid * D_ + c] = y;
    if (outbf) outbf[(size_t)wid * D_ + c] = f2bf(y);
  }
}

// ---- causal context window gather: xctx[((bs,p),c),d] ----
__global__ void k_window(float* __restrict__ xctx, const float* __restrict__ tok2) {
  int i = blockIdx.x * blockDim.x + threadIdx.x;   // < 65536*256
  int d = i & 255, row = i >> 8;
  int c = row & 3, bp = row >> 2;
  int p = bp & 63, bs = bp >> 6;
  int s = bs & 31, b = bs >> 5;
  int ti = s + c - (CTX_ - 1);
  xctx[i] = (ti >= 0) ? tok2[((size_t)((b * S_ + ti) * P_ + p) << 8) + d] : 0.f;
}

// ---- scan seed: ctx_tok[(b,0,p)] = cur[(b,0,p)] (= xctx last window slot) ----
__global__ void k_copy_s0(float* __restrict__ ctx_tok, const float* __restrict__ xctx) {
  int i = blockIdx.x * blockDim.x + threadIdx.x;   // < 512*256
  int d = i & 255, r = i >> 8;
  int b = r >> 6, p = r & 63;
  ctx_tok[((size_t)(b * S_ * P_ + p) << 8) + d] =
      xctx[((size_t)(((b * S_ + 0) * P_ + p) * CTX_ + (CTX_ - 1)) << 8) + d];
}

// ---- build concat([cur_s, cur_s - prev]) as bf16 for the delta GEMM ----
__global__ void k_concat(u16* __restrict__ cat, const float* __restrict__ xctx,
                         const float* __restrict__ ctx_tok, int s) {
  int i = blockIdx.x * blockDim.x + threadIdx.x;   // < 512*256
  int d = i & 255, r = i >> 8;
  int b = r >> 6, p = r & 63;
  float cv = xctx[((size_t)(((b * S_ + s) * P_ + p) * CTX_ + (CTX_ - 1)) << 8) + d];
  float pv = ctx_tok[((size_t)((b * S_ + (s - 1)) * P_ + p) << 8) + d];
  cat[(size_t)r * (2 * D_) + d]      = f2bf(cv);
  cat[(size_t)r * (2 * D_) + D_ + d] = f2bf(cv - pv);
}

// ---- codebook squared norms ----
__global__ void k_cbnorm(float* __restrict__ cbn, const float* __restrict__ cb) {
  int j = blockIdx.x * blockDim.x + threadIdx.x;
  if (j >= KCB_) return;
  float s = 0.f;
  for (int d = 0; d < D_; ++d) { float v = cb[j * D_ + d]; s += v * v; }
  cbn[j] = s;
}

// ---- argmin_j ( |cb_j|^2 - 2*dot[r][j] ), one wave per row ----
__global__ void k_argmin(int* __restrict__ idx, int* __restrict__ idx_out,
                         const float* __restrict__ dot, const float* __restrict__ cbn,
                         int rows) {
  int wid  = (blockIdx.x * blockDim.x + threadIdx.x) >> 5;
  int lane = threadIdx.x & 31;
  if (wid >= rows) return;
  const float* dr = dot + (size_t)wid * KCB_;
  float best = 3.4e38f; int bi = 0x7fffffff;
  for (int j = lane; j < KCB_; j += 32) {
    float v = cbn[j] - 2.f * dr[j];
    if (v < best || (v == best && j < bi)) { best = v; bi = j; }
  }
#pragma unroll
  for (int o = 16; o >= 1; o >>= 1) {
    float ob = __shfl_xor(best, o, 32);
    int   oi = __shfl_xor(bi,   o, 32);
    if (ob < best || (ob == best && oi < bi)) { best = ob; bi = oi; }
  }
  if (lane == 0) { idx[wid] = bi; idx_out[wid] = bi; }
}

// ---- q gather ----
__global__ void k_gather_q(float* __restrict__ q, const float* __restrict__ cb,
                           const int* __restrict__ idx) {
  int i = blockIdx.x * blockDim.x + threadIdx.x;   // < 16384*256
  int d = i & 255, r = i >> 8;
  q[i] = cb[(size_t)idx[r] * D_ + d];
}

// ---- deterministic two-stage sq-diff reduction ----
__global__ void k_sqdiff(float* __restrict__ partials, const float* __restrict__ a,
                         const float* __restrict__ b, int n) {
  __shared__ float sm[256];
  float acc = 0.f;
  for (int i = blockIdx.x * blockDim.x + threadIdx.x; i < n; i += gridDim.x * blockDim.x) {
    float d = a[i] - b[i];
    acc += d * d;
  }
  sm[threadIdx.x] = acc;
  __syncthreads();
  for (int o = 128; o > 0; o >>= 1) {
    if (threadIdx.x < o) sm[threadIdx.x] += sm[threadIdx.x + o];
    __syncthreads();
  }
  if (threadIdx.x == 0) partials[blockIdx.x] = sm[0];
}
__global__ void k_finalize(float* __restrict__ scal, const float* __restrict__ partials,
                           int np, float invN) {
  float s = 0.f;
  for (int i = 0; i < np; ++i) s += partials[i];
  float msd = s * invN;
  scal[0] = 0.25f * msd;   // commit
  scal[1] = msd;           // cb_loss
  scal[2] = 1.25f * msd;   // total
}

extern "C" void kernel_launch(void* const* d_in, const int* in_sizes, int n_in,
                              void* d_out, int out_size, void* d_ws, size_t ws_size,
                              hipStream_t stream) {
  (void)in_sizes; (void)n_in; (void)out_size; (void)ws_size;

  const float* obs       = (const float*)d_in[0];
  const float* patch_w   = (const float*)d_in[2];
  const float* patch_b   = (const float*)d_in[3];
  const float* pos_embed = (const float*)d_in[4];
  const float* sa_qkv_w  = (const float*)d_in[5];
  const float* sa_qkv_b  = (const float*)d_in[6];
  const float* sa_out_w  = (const float*)d_in[7];
  const float* sa_out_b  = (const float*)d_in[8];
  const float* sa_ln_g   = (const float*)d_in[9];
  const float* sa_ln_b   = (const float*)d_in[10];
  const float* ctx_qkv_w = (const float*)d_in[11];
  const float* ctx_qkv_b = (const float*)d_in[12];
  const float* ctx_out_w = (const float*)d_in[13];
  const float* ctx_out_b = (const float*)d_in[14];
  const float* ctx_ff1_w = (const float*)d_in[15];
  const float* ctx_ff1_b = (const float*)d_in[16];
  const float* ctx_ff2_w = (const float*)d_in[17];
  const float* ctx_ff2_b = (const float*)d_in[18];
  const float* ctx_ln1_g = (const float*)d_in[19];
  const float* ctx_ln1_b = (const float*)d_in[20];
  const float* ctx_ln2_g = (const float*)d_in[21];
  const float* ctx_ln2_b = (const float*)d_in[22];
  const float* delta_w   = (const float*)d_in[23];
  const float* delta_b   = (const float*)d_in[24];
  const float* codebook  = (const float*)d_in[25];
  const float* dec_w1    = (const float*)d_in[26];
  const float* dec_b1    = (const float*)d_in[27];
  const float* dec_w2    = (const float*)d_in[28];
  const float* dec_b2    = (const float*)d_in[29];
  const float* dec_w3    = (const float*)d_in[30];
  const float* dec_b3    = (const float*)d_in[31];

  float* recon   = (float*)d_out;                               // 16384 x 64
  int*   idx_out = (int*)((float*)d_out + (size_t)B_*S_*P_*OBS_);
  float* scal    = (float*)d_out + (size_t)B_*S_*P_*OBS_ + (size_t)B_*S_*P_;

  const int T1 = B_ * S_ * P_;        // 16384 tokens (SA)
  const int T2 = T1 * CTX_;           // 65536 tokens (ctx)

  // ---- workspace carve-out ----
  char* wsb = (char*)d_ws;
  size_t cur = 0;
  auto alloc = [&](size_t bytes) -> void* {
    cur = (cur + 255) & ~(size_t)255;
    void* p = wsb + cur;
    cur += bytes;
    return p;
  };
  u16* obs_bf       = (u16*)alloc((size_t)B_*S_*OBS_ * 2);
  u16* patch_w_bf   = (u16*)alloc((size_t)D_*OBS_ * 2);
  u16* sa_qkv_w_bf  = (u16*)alloc((size_t)3*D_*D_ * 2);
  u16* sa_out_w_bf  = (u16*)alloc((size_t)D_*D_ * 2);
  u16* ctx_qkv_w_bf = (u16*)alloc((size_t)2*3*D_*D_ * 2);
  u16* ctx_out_w_bf = (u16*)alloc((size_t)2*D_*D_ * 2);
  u16* ctx_ff1_w_bf = (u16*)alloc((size_t)2*H_*D_ * 2);
  u16* ctx_ff2_w_bf = (u16*)alloc((size_t)2*D_*H_ * 2);
  u16* delta_w_bf   = (u16*)alloc((size_t)D_*2*D_ * 2);
  u16* codebook_bf  = (u16*)alloc((size_t)KCB_*D_ * 2);
  u16* dec_w1_bf    = (u16*)alloc((size_t)H_*D_ * 2);
  u16* dec_w2_bf    = (u16*)alloc((size_t)H_*H_ * 2);
  u16* dec_w3_bf    = (u16*)alloc((size_t)OBS_*H_ * 2);

  float* tokA     = (float*)alloc((size_t)B_*S_*D_ * 4);
  float* tok      = (float*)alloc((size_t)T1*D_ * 4);
  float* tok2     = (float*)alloc((size_t)T1*D_ * 4);
  float* xctx     = (float*)alloc((size_t)T2*D_ * 4);
  float* ctx_tok  = (float*)alloc((size_t)T1*D_ * 4);
  float* qbuf     = (float*)alloc((size_t)T1*D_ * 4);
  u16*   ctok_bf  = (u16*)alloc((size_t)T1*D_ * 2);
  int*   idx_ws   = (int*)alloc((size_t)T1 * 4);
  float* cbn      = (float*)alloc((size_t)KCB_ * 4);
  float* partials = (float*)alloc((size_t)4096 * 4);
  u16*   cat_bf   = (u16*)alloc((size_t)(B_*P_)*(2*D_) * 2);

  u16*   xbf     = (u16*)alloc((size_t)T2*D_ * 2);        // 32 MB
  u16*   qkv_bf  = (u16*)alloc((size_t)T2*3*D_ * 2);      // 96 MB
  u16*   attn_bf = (u16*)alloc((size_t)T2*D_ * 2);        // 32 MB
  float* proj    = (float*)alloc((size_t)T2*D_ * 4);      // 64 MB (also ff2 out)
  u16*   hbf     = (u16*)alloc((size_t)T2*H_ * 2);        // 128 MB

  // phase-exclusive aliases
  float* dot   = (float*)hbf;      // 16384 x 1024 f32 (VQ)
  u16*   h1bf  = attn_bf;          // 16384 x 1024 bf16
  u16*   h2bf  = xbf;              // 16384 x 1024 bf16
  u16*   q_bf  = qkv_bf;           // 16384 x 256  bf16

  auto cvt = [&](u16* dst, const float* src, int n) {
    k_cvt<<<dim3((n + 255) / 256), dim3(256), 0, stream>>>(dst, src, n);
  };
  auto gemm = [&](const u16* A, int lda, const u16* W, int ldb, const float* bias,
                  int M, int N, int K, float* oF, u16* oB, int ldc, bool relu,
                  int rpg = -1, long long gs = 0, long long ro = 0) {
    if (rpg < 0) { rpg = M; gs = 0; ro = 0; }
    k_gemm<<<dim3(N / 64, M / 128), dim3(128), 0, stream>>>(
        A, lda, W, ldb, bias, K, oF, oB, ldc, relu ? 1 : 0, rpg, gs, ro);
  };

  // ---- weight / input bf16 conversion ----
  cvt(obs_bf, obs, B_*S_*OBS_);
  cvt(patch_w_bf,   patch_w,   D_*OBS_);
  cvt(sa_qkv_w_bf,  sa_qkv_w,  3*D_*D_);
  cvt(sa_out_w_bf,  sa_out_w,  D_*D_);
  cvt(ctx_qkv_w_bf, ctx_qkv_w, 2*3*D_*D_);
  cvt(ctx_out_w_bf, ctx_out_w, 2*D_*D_);
  cvt(ctx_ff1_w_bf, ctx_ff1_w, 2*H_*D_);
  cvt(ctx_ff2_w_bf, ctx_ff2_w, 2*D_*H_);
  cvt(delta_w_bf,   delta_w,   D_*2*D_);
  cvt(codebook_bf,  codebook,  KCB_*D_);
  cvt(dec_w1_bf,    dec_w1,    H_*D_);
  cvt(dec_w2_bf,    dec_w2,    H_*H_);
  cvt(dec_w3_bf,    dec_w3,    OBS_*H_);

  // ---- stage 1: patch projection + pos embed ----
  gemm(obs_bf, OBS_, patch_w_bf, OBS_, patch_b, B_*S_, D_, OBS_, tokA, nullptr, D_, false);
  k_expand<<<dim3(T1*D_/256), dim3(256), 0, stream>>>(tok, tokA, pos_embed);

  // ---- stage 2: self-attention over P=64 ----
  cvt(xbf, tok, T1*D_);
  gemm(xbf, D_, sa_qkv_w_bf, D_, sa_qkv_b, T1, 3*D_, D_, nullptr, qkv_bf, 3*D_, false);
  k_attn<64><<<dim3((B_*S_*NH_*64)/256), dim3(256), 0, stream>>>(qkv_bf, attn_bf, B_*S_, 0.125f);
  gemm(attn_bf, D_, sa_out_w_bf, D_, sa_out_b, T1, D_, D_, proj, nullptr, D_, false);
  k_add_ln<<<dim3(T1/8), dim3(256), 0, stream>>>(proj, tok, sa_ln_g, sa_ln_b, tok2, nullptr, T1);

  // ---- stage 3: causal window build ----
  k_window<<<dim3(T2*D_/256), dim3(256), 0, stream>>>(xctx, tok2);
  cvt(xbf, xctx, T2*D_);

  // ---- stage 4: two ctx transformer layers (L=4) ----
  for (int i = 0; i < 2; ++i) {
    gemm(xbf, D_, ctx_qkv_w_bf + (size_t)i*3*D_*D_, D_, ctx_qkv_b + i*3*D_,
         T2, 3*D_, D_, nullptr, qkv_bf, 3*D_, false);
    k_attn<4><<<dim3((T1*NH_*4)/256), dim3(256), 0, stream>>>(qkv_bf, attn_bf, T1, 0.125f);
    gemm(attn_bf, D_, ctx_out_w_bf + (size_t)i*D_*D_, D_, ctx_out_b + i*D_,
         T2, D_, D_, proj, nullptr, D_, false);
    k_add_ln<<<dim3(T2/8), dim3(256), 0, stream>>>(xctx, proj, ctx_ln1_g + i*D_,
                                                   ctx_ln1_b + i*D_, xctx, xbf, T2);
    gemm(xbf, D_, ctx_ff1_w_bf + (size_t)i*H_*D_, D_, ctx_ff1_b + i*H_,
         T2, H_, D_, nullptr, hbf, H_, true);
    gemm(hbf, H_, ctx_ff2_w_bf + (size_t)i*D_*H_, H_, ctx_ff2_b + i*D_,
         T2, D_, H_, proj, nullptr, D_, false);
    k_add_ln<<<dim3(T2/8), dim3(256), 0, stream>>>(xctx, proj, ctx_ln2_g + i*D_,
                                                   ctx_ln2_b + i*D_, xctx, xbf, T2);
  }

  // ---- stage 5: sequential delta scan over S ----
  k_copy_s0<<<dim3((B_*P_*D_)/256), dim3(256), 0, stream>>>(ctx_tok, xctx);
  for (int s = 1; s < S_; ++s) {
    k_concat<<<dim3((B_*P_*D_)/256), dim3(256), 0, stream>>>(cat_bf, xctx, ctx_tok, s);
    gemm(cat_bf, 2*D_, delta_w_bf, 2*D_, delta_b, B_*P_, D_, 2*D_,
         ctx_tok, nullptr, D_, false,
         /*rpg=*/P_, /*gStride=*/(long long)S_*P_, /*rOff=*/(long long)s*P_);
  }

  // ---- stage 6: VQ ----
  cvt(ctok_bf, ctx_tok, T1*D_);
  k_cbnorm<<<dim3(KCB_/256), dim3(256), 0, stream>>>(cbn, codebook);
  gemm(ctok_bf, D_, codebook_bf, D_, nullptr, T1, KCB_, D_, dot, nullptr, KCB_, false);
  k_argmin<<<dim3(T1/8), dim3(256), 0, stream>>>(idx_ws, idx_out, dot, cbn, T1);
  k_gather_q<<<dim3(T1*D_/256), dim3(256), 0, stream>>>(qbuf, codebook, idx_ws);
  k_sqdiff<<<dim3(4096), dim3(256), 0, stream>>>(partials, ctx_tok, qbuf, T1*D_);
  k_finalize<<<dim3(1), dim3(1), 0, stream>>>(scal, partials, 4096, 1.f/((float)T1*(float)D_));

  // ---- stage 7: decoder (q_st == q numerically) ----
  cvt(q_bf, qbuf, T1*D_);
  gemm(q_bf, D_, dec_w1_bf, D_, dec_b1, T1, H_, D_, nullptr, h1bf, H_, true);
  gemm(h1bf, H_, dec_w2_bf, H_, dec_b2, T1, H_, H_, nullptr, h2bf, H_, true);
  gemm(h2bf, H_, dec_w3_bf, H_, dec_b3, T1, OBS_, H_, recon, nullptr, OBS_, false);
}
